// AutoregressiveTransformer_39058432590360
// MI455X (gfx1250) — matrix operand order repaired and checked by reading, not compiled, not run on hardware
//
#include <hip/hip_runtime.h>
#include <hip/hip_bf16.h>
#include <math.h>

typedef __attribute__((ext_vector_type(16))) _Float16 v16h;
typedef __attribute__((ext_vector_type(8)))  _Float16 v8h;
typedef __attribute__((ext_vector_type(8)))  float    v8f;

#define NB      16          // batch
#define GRID_W  80
#define LPOS    6400        // 80*80
#define DM      768
#define ROW     6581        // per-batch output row length
#define K1      8064        // conv1 GEMM K (reordered: 768*9 broadcast part + 128*9 map part)
#define K1_OUT  6912        // 768*9
#define K2      576         // conv2 GEMM K (9*64)
#define PI_F    3.14159265358979f

// ---- workspace byte offsets (all 32B-aligned where vector loads need it) ----
#define WS_OUT_H   0u          // 16*768 f16                      (24576 B)
#define WS_WP1     32768u      // conv1 weights f16 [64][8064]    (1032192 B)
#define WS_WP2     1064960u    // conv2 weights f16 [64][576]     (73728 B)
#define WS_X1H     1138688u    // bn1+relu(conv1) f16 [b][oc][pos] (13107200 B)
#define WS_X1P     14245888u   // maxpool f16 [b][pos][oc]         (13107200 B)
#define WS_SMALL   27353088u   // floats: lf[16*128], f2[16*1088]

__device__ inline v8f wmma_f16_acc(v16h a, v16h b, v8f c) {
  // D = A(16x32) * B(32x16) + C, f32 accumulate
  return __builtin_amdgcn_wmma_f32_16x16x32_f16(false, a, false, b, (short)0, c, false, false);
}
__device__ inline v16h make_a(v8h r0, v8h r1) {
  v16h a;
  for (int j = 0; j < 8; ++j) { a[j] = r0[j]; a[8 + j] = r1[j]; }
  return a;
}
__device__ inline float sigmoidf_(float x) { return 1.f / (1.f + expf(-x)); }

// ======================= pack: fp32 -> f16 conversions =======================
__global__ __launch_bounds__(256)
void k_pack(const float* __restrict__ output_f,
            const float* __restrict__ conv1_w,   // [64][896][3][3]
            const float* __restrict__ conv2_w,   // [64][64][3][3]
            _Float16* __restrict__ out_h,
            _Float16* __restrict__ wp1,
            _Float16* __restrict__ wp2)
{
  int idx = blockIdx.x * 256 + threadIdx.x;
  const int N0 = NB * DM;            // 12288
  const int N1 = 64 * K1;            // 516096
  const int N2 = 64 * K2;            // 36864
  if (idx < N0) {
    out_h[idx] = (_Float16)output_f[idx];
  } else if (idx < N0 + N1) {
    int j  = idx - N0;
    int oc = j / K1;
    int k  = j % K1;
    int tap, ci;
    if (k < K1_OUT) { tap = k / 768; ci = k % 768; }
    else            { int km = k - K1_OUT; tap = km >> 7; ci = 768 + (km & 127); }
    int dy = tap / 3, dx = tap % 3;
    wp1[j] = (_Float16)conv1_w[(((size_t)oc * 896 + ci) * 3 + dy) * 3 + dx];
  } else if (idx < N0 + N1 + N2) {
    int j  = idx - N0 - N1;
    int oc = j / K2;
    int k  = j % K2;
    int tap = k >> 6, ci = k & 63;
    int dy = tap / 3, dx = tap % 3;
    wp2[j] = (_Float16)conv2_w[(((size_t)oc * 64 + ci) * 3 + dy) * 3 + dx];
  }
}

// ======================= conv1: implicit GEMM via WMMA =======================
// block = 64 thr (2 waves); wave w computes oc tiles [32w,32w+16) and
// [32w+16,32w+32) for one 16-position x-tile, reusing the A fragment and
// alternating two accumulators (hides WMMA->WMMA RAW hazard NOPs).
// K loop restructured per-tap: no runtime div/mod, branch-free zero padding
// via pointer select into a zeroed LDS mirror.
__global__ __launch_bounds__(64)
void k_conv1(const float* __restrict__ map_f,
             const float* __restrict__ conv1_b,
             const float* __restrict__ bn1_g, const float* __restrict__ bn1_b,
             const float* __restrict__ bn1_m, const float* __restrict__ bn1_v,
             const _Float16* __restrict__ out_h,
             const _Float16* __restrict__ wp1,
             _Float16* __restrict__ x1h)
{
  __shared__ __align__(16) _Float16 s_outz[2 * 768];   // [0]=output_f, [1]=zeros
  __shared__ __align__(16) _Float16 s_map[3 * 18 * 128];

  const int b      = blockIdx.y;
  const int m_base = blockIdx.x * 16;           // 80 % 16 == 0: tile stays in one row
  const int y      = m_base / GRID_W;
  const int x_base = m_base % GRID_W;
  const int tid    = threadIdx.x;

  for (int i = tid; i < 2 * 768 + 3 * 18 * 128; i += 64) {
    if (i < 768) {
      s_outz[i] = out_h[b * DM + i];
    } else if (i < 2 * 768) {
      s_outz[i] = (_Float16)0.f;                // zero mirror for OOB taps
    } else {
      int j   = i - 2 * 768;
      int c   = j & 127;
      int col = (j >> 7) % 18;
      int row = (j >> 7) / 18;
      int gy = y - 1 + row, gx = x_base - 1 + col;
      _Float16 v = (_Float16)0.f;
      if (gy >= 0 && gy < GRID_W && gx >= 0 && gx < GRID_W)
        v = (_Float16)map_f[((size_t)(b * LPOS + gy * GRID_W + gx)) * 128 + c];
      s_map[j] = v;
    }
  }
  __syncthreads();

  const int lane = tid & 31;
  const int wave = tid >> 5;
  const int half = lane >> 4;                   // K-half select
  const int lm   = lane & 15;                   // M row (A) / N col (B)
  const int xg   = x_base + lm;
  const int oc0  = wave * 32 + lm;              // first oc tile
  const _Float16* wrow0 = wp1 + (size_t)oc0 * K1;
  const _Float16* wrow1 = wp1 + (size_t)(oc0 + 16) * K1;
  const int aoff0 = half * 16;                  // A run offsets within 32-chunk
  const int boff  = half * 16;                  // B run offset

  v8f acc0, acc1;
  for (int r = 0; r < 8; ++r) { acc0[r] = 0.f; acc1[r] = 0.f; }

  // ---- phase A: broadcast output_f channels (K = 9 taps x 768) ----
  for (int tap = 0; tap < 9; ++tap) {
    int gy = y + tap / 3 - 1;
    int gx = xg + tap % 3 - 1;
    bool ok = (gy >= 0) & (gy < GRID_W) & (gx >= 0) & (gx < GRID_W);
    const _Float16* ab  = ok ? s_outz : (s_outz + 768);   // branch-free padding
    const _Float16* wb0 = wrow0 + tap * 768;
    const _Float16* wb1 = wrow1 + tap * 768;
#pragma unroll 4
    for (int c0 = 0; c0 < 768; c0 += 32) {
      v8h r0 = *(const v8h*)(ab + c0 + aoff0);
      v8h r1 = *(const v8h*)(ab + c0 + 8 + aoff0);
      v16h a  = make_a(r0, r1);
      v16h b0 = *(const v16h*)(wb0 + c0 + boff);
      v16h b1 = *(const v16h*)(wb1 + c0 + boff);
      acc0 = wmma_f16_acc(a, b0, acc0);
      acc1 = wmma_f16_acc(a, b1, acc1);
    }
  }

  // ---- phase B: map_f channels (K = 9 taps x 128), patch already padded ----
  for (int tap = 0; tap < 9; ++tap) {
    int dy = tap / 3, dx = tap % 3;
    const _Float16* ab  = s_map + ((dy * 18 + (lm + dx)) << 7);
    const _Float16* wb0 = wrow0 + K1_OUT + tap * 128;
    const _Float16* wb1 = wrow1 + K1_OUT + tap * 128;
#pragma unroll
    for (int c0 = 0; c0 < 128; c0 += 32) {
      v8h r0 = *(const v8h*)(ab + c0 + aoff0);
      v8h r1 = *(const v8h*)(ab + c0 + 8 + aoff0);
      v16h a  = make_a(r0, r1);
      v16h b0 = *(const v16h*)(wb0 + c0 + boff);
      v16h b1 = *(const v16h*)(wb1 + c0 + boff);
      acc0 = wmma_f16_acc(a, b0, acc0);
      acc1 = wmma_f16_acc(a, b1, acc1);
    }
  }

  // fused bias + BN1 + ReLU, f16 store [b][oc][pos]
  for (int tnum = 0; tnum < 2; ++tnum) {
    int oc = oc0 + tnum * 16;
    float sc = bn1_g[oc] * rsqrtf(bn1_v[oc] + 1e-5f);
    float sh = bn1_b[oc] - bn1_m[oc] * sc;
    float cb = conv1_b[oc];
    v8h ov;
    for (int r = 0; r < 8; ++r) {
      float v = ((tnum ? acc1[r] : acc0[r]) + cb) * sc + sh;
      ov[r] = (_Float16)(v > 0.f ? v : 0.f);
    }
    *(v8h*)(x1h + ((size_t)(b * 64 + oc)) * LPOS + m_base + half * 8) = ov;
  }
}

// ========================== maxpool 3x3 stride 1 =============================
// x1h [b][oc][pos] -> x1p [b][pos][oc] (channel-fast for conv2 A loads)
__global__ __launch_bounds__(256)
void k_maxpool(const _Float16* __restrict__ x1h, _Float16* __restrict__ x1p)
{
  int tid = blockIdx.x * 256 + threadIdx.x;
  if (tid >= NB * 64 * LPOS) return;
  int pos = tid % LPOS;
  int c   = (tid / LPOS) & 63;
  int b   = tid / (LPOS * 64);
  int y = pos / GRID_W, x = pos % GRID_W;
  const _Float16* src = x1h + ((size_t)(b * 64 + c)) * LPOS;
  float m = -INFINITY;
  for (int dy = -1; dy <= 1; ++dy)
    for (int dx = -1; dx <= 1; ++dx) {
      int gy = y + dy, gx = x + dx;
      if (gy >= 0 && gy < GRID_W && gx >= 0 && gx < GRID_W) {
        float v = (float)src[gy * GRID_W + gx];
        m = v > m ? v : m;
      }
    }
  x1p[((size_t)(b * LPOS + pos)) * 64 + c] = (_Float16)m;
}

// ================= conv2 WMMA + fused BN2/ReLU + conv3(1x1) ==================
// Same tiling as conv1: 64 thr, 2 waves, 2 oc-tiles per wave, zero-padded
// LDS input patch so the K loop is branch-free.
__global__ __launch_bounds__(64)
void k_conv2(const _Float16* __restrict__ x1p,
             const _Float16* __restrict__ wp2,
             const float* __restrict__ conv2_b,
             const float* __restrict__ bn2_g, const float* __restrict__ bn2_b,
             const float* __restrict__ bn2_m, const float* __restrict__ bn2_v,
             const float* __restrict__ conv3_w, const float* __restrict__ conv3_b,
             float* __restrict__ out)
{
  __shared__ __align__(16) _Float16 s_p[3 * 18 * 64];  // zero-padded input patch
  __shared__ float s_tile[16 * 64];                    // [m][oc] for conv3 fusion

  const int b      = blockIdx.y;
  const int m_base = blockIdx.x * 16;
  const int y      = m_base / GRID_W;
  const int x_base = m_base % GRID_W;
  const int tid    = threadIdx.x;

  for (int j = tid; j < 3 * 18 * 64; j += 64) {
    int c   = j & 63;
    int col = (j >> 6) % 18;
    int row = (j >> 6) / 18;
    int gy = y - 1 + row, gx = x_base - 1 + col;
    _Float16 v = (_Float16)0.f;
    if (gy >= 0 && gy < GRID_W && gx >= 0 && gx < GRID_W)
      v = x1p[((size_t)(b * LPOS + gy * GRID_W + gx)) * 64 + c];
    s_p[j] = v;
  }
  __syncthreads();

  const int lane = tid & 31;
  const int wave = tid >> 5;
  const int half = lane >> 4;
  const int lm   = lane & 15;
  const int oc0  = wave * 32 + lm;
  const _Float16* wrow0 = wp2 + (size_t)oc0 * K2;
  const _Float16* wrow1 = wp2 + (size_t)(oc0 + 16) * K2;
  const int aoff0 = half * 16;
  const int boff  = half * 16;

  v8f acc0, acc1;
  for (int r = 0; r < 8; ++r) { acc0[r] = 0.f; acc1[r] = 0.f; }

  for (int tap = 0; tap < 9; ++tap) {
    int dy = tap / 3, dx = tap % 3;
    const _Float16* ab  = s_p + ((dy * 18 + (lm + dx)) << 6);
    const _Float16* wb0 = wrow0 + tap * 64;
    const _Float16* wb1 = wrow1 + tap * 64;
#pragma unroll
    for (int c0 = 0; c0 < 64; c0 += 32) {
      v8h r0 = *(const v8h*)(ab + c0 + aoff0);
      v8h r1 = *(const v8h*)(ab + c0 + 8 + aoff0);
      v16h a  = make_a(r0, r1);
      v16h b0 = *(const v16h*)(wb0 + c0 + boff);
      v16h b1 = *(const v16h*)(wb1 + c0 + boff);
      acc0 = wmma_f16_acc(a, b0, acc0);
      acc1 = wmma_f16_acc(a, b1, acc1);
    }
  }

  for (int tnum = 0; tnum < 2; ++tnum) {
    int oc = oc0 + tnum * 16;
    float sc = bn2_g[oc] * rsqrtf(bn2_v[oc] + 1e-5f);
    float sh = bn2_b[oc] - bn2_m[oc] * sc;
    float cb = conv2_b[oc];
    for (int r = 0; r < 8; ++r) {
      float v = ((tnum ? acc1[r] : acc0[r]) + cb) * sc + sh;
      s_tile[(r + half * 8) * 64 + oc] = v > 0.f ? v : 0.f;
    }
  }
  __syncthreads();

  if (tid < 16) {                               // conv3 1x1, 64 -> 1
    float d = conv3_b[0];
    for (int c = 0; c < 64; ++c) d += conv3_w[c] * s_tile[tid * 64 + c];
    out[(size_t)b * ROW + m_base + tid] = d;    // loc_logits
  }
}

// =================== per-batch argmax + location_f gather ====================
__global__ __launch_bounds__(256)
void k_argmax(const float* __restrict__ out, const float* __restrict__ map_f,
              float* __restrict__ lf)
{
  __shared__ float sv[256];
  __shared__ int   si[256];
  int b = blockIdx.x, t = threadIdx.x;
  const float* row = out + (size_t)b * ROW;
  float best = -INFINITY; int bi = 0;
  for (int i = t; i < LPOS; i += 256) {
    float v = row[i];
    if (v > best) { best = v; bi = i; }
  }
  sv[t] = best; si[t] = bi; __syncthreads();
  for (int s = 128; s > 0; s >>= 1) {
    if (t < s) {
      if (sv[t + s] > sv[t] || (sv[t + s] == sv[t] && si[t + s] < si[t])) {
        sv[t] = sv[t + s]; si[t] = si[t + s];
      }
    }
    __syncthreads();
  }
  int loc = si[0];
  if (t < 128) lf[b * 128 + t] = map_f[((size_t)(b * LPOS + loc)) * 128 + t];
}

// ============ heads part 1: wl/th MLPs, mixtures, PE, f2 assembly ============
__global__ __launch_bounds__(128)
void k_heads1(const float* __restrict__ output_f, const float* __restrict__ lf,
              const float* __restrict__ wl_w1, const float* __restrict__ wl_b1,
              const float* __restrict__ wl_w2, const float* __restrict__ wl_b2,
              const float* __restrict__ th_w1, const float* __restrict__ th_b1,
              const float* __restrict__ th_w2, const float* __restrict__ th_b2,
              float* __restrict__ out, float* __restrict__ f2w)
{
  __shared__ float s_lf[128], s_h[128], s_wl[50], s_th[40], s_thmu[10], s_pv[3];
  int b = blockIdx.x, t = threadIdx.x;
  s_lf[t] = lf[b * 128 + t];
  __syncthreads();

  { float a = wl_b1[t]; for (int i = 0; i < 128; ++i) a += s_lf[i] * wl_w1[i * 128 + t];
    s_h[t] = a > 0.f ? a : 0.f; }
  __syncthreads();
  if (t < 50) { float a = wl_b2[t]; for (int i = 0; i < 128; ++i) a += s_h[i] * wl_w2[i * 50 + t];
    s_wl[t] = a; }
  __syncthreads();
  { float a = th_b1[t]; for (int i = 0; i < 128; ++i) a += s_lf[i] * th_w1[i * 128 + t];
    s_h[t] = a > 0.f ? a : 0.f; }
  __syncthreads();
  if (t < 40) { float a = th_b2[t]; for (int i = 0; i < 128; ++i) a += s_h[i] * th_w2[i * 40 + t];
    s_th[t] = a; }
  __syncthreads();

  float* row = out + (size_t)b * ROW;
  if (t < 10) {
    row[6400 + t] = s_wl[t];                                   // wl_logits
    float mu0 = s_wl[10 + t * 4], mu1 = s_wl[10 + t * 4 + 1];
    row[6410 + t * 2]     = mu0;                               // wl_mu
    row[6410 + t * 2 + 1] = mu1;
    row[6430 + t * 2]     = 0.5f * sigmoidf_(fmaxf(s_wl[10 + t * 4 + 2], -5.f)); // wl_sig
    row[6430 + t * 2 + 1] = 0.5f * sigmoidf_(fmaxf(s_wl[10 + t * 4 + 3], -5.f));
    row[6450 + t] = s_th[t];                                   // th_logits
    float c0 = s_th[10 + t * 3], s0 = s_th[10 + t * 3 + 1], r2 = s_th[10 + t * 3 + 2];
    float kap = 7.f + expf(fminf(r2, 5.f));
    float nrm = sqrtf(c0 * c0 + s0 * s0) + 1e-3f;
    float cn = c0 / nrm, sn = s0 / nrm;
    float mu = asinf(fminf(fmaxf(sn, -1.f), 1.f));
    if (cn < 0.f) {
      float sg = (sn > 0.f) ? 1.f : ((sn < 0.f) ? -1.f : 0.f);
      mu = sg * PI_F - mu;
    }
    row[6460 + t] = cn; row[6470 + t] = sn;
    row[6480 + t] = kap; row[6490 + t] = mu;
    s_thmu[t] = mu;
  }
  __syncthreads();
  if (t == 0) {
    int kwl = 0; float bv = s_wl[0];
    for (int i = 1; i < 10; ++i) if (s_wl[i] > bv) { bv = s_wl[i]; kwl = i; }
    s_pv[0] = expf(s_wl[10 + kwl * 4]);
    s_pv[1] = expf(s_wl[10 + kwl * 4 + 1]);
    int kth = 0; float bt = s_th[0];
    for (int i = 1; i < 10; ++i) if (s_th[i] > bt) { bt = s_th[i]; kth = i; }
    s_pv[2] = s_thmu[kth];
  }
  __syncthreads();

  float* f2 = f2w + (size_t)b * 1088;
  if (t < 96) {                                 // positional encoding -> f2[896:1088]
    int ti = t >> 5, i = t & 31;
    float ang = s_pv[ti] * exp2f((float)i) * PI_F;
    f2[896 + ti * 64 + i]      = sinf(ang);
    f2[896 + ti * 64 + 32 + i] = cosf(ang);
  }
  for (int i = t; i < 768; i += 128) f2[i] = output_f[b * DM + i];
  f2[768 + t] = s_lf[t];
}

// ================= heads part 2: mv / s / om MLPs + mixtures =================
__global__ __launch_bounds__(128)
void k_heads2(const float* __restrict__ f2w,
              const float* __restrict__ mv_w1, const float* __restrict__ mv_b1,
              const float* __restrict__ mv_w2, const float* __restrict__ mv_b2,
              const float* __restrict__ s_w1,  const float* __restrict__ s_b1,
              const float* __restrict__ s_w2,  const float* __restrict__ s_b2,
              const float* __restrict__ om_w1, const float* __restrict__ om_b1,
              const float* __restrict__ om_w2, const float* __restrict__ om_b2,
              float* __restrict__ out)
{
  __shared__ float s_f2[1088], s_h[128], s_so[30], s_om[40];
  int b = blockIdx.x, t = threadIdx.x;
  for (int i = t; i < 1088; i += 128) s_f2[i] = f2w[(size_t)b * 1088 + i];
  __syncthreads();
  float* row = out + (size_t)b * ROW;

  { float a = mv_b1[t]; for (int i = 0; i < 1088; ++i) a += s_f2[i] * mv_w1[i * 128 + t];
    s_h[t] = a > 0.f ? a : 0.f; }
  __syncthreads();
  if (t == 0) {
    float a = mv_b2[0];
    for (int i = 0; i < 128; ++i) a += s_h[i] * mv_w2[i];
    row[6500] = a;                              // moving
  }
  __syncthreads();

  { float a = s_b1[t]; for (int i = 0; i < 1088; ++i) a += s_f2[i] * s_w1[i * 128 + t];
    s_h[t] = a > 0.f ? a : 0.f; }
  __syncthreads();
  if (t < 30) { float a = s_b2[t]; for (int i = 0; i < 128; ++i) a += s_h[i] * s_w2[i * 30 + t];
    s_so[t] = a; }
  __syncthreads();

  { float a = om_b1[t]; for (int i = 0; i < 1088; ++i) a += s_f2[i] * om_w1[i * 128 + t];
    s_h[t] = a > 0.f ? a : 0.f; }
  __syncthreads();
  if (t < 40) { float a = om_b2[t]; for (int i = 0; i < 128; ++i) a += s_h[i] * om_w2[i * 40 + t];
    s_om[t] = a; }
  __syncthreads();

  if (t < 10) {
    row[6501 + t] = s_so[t];                                   // s_logits
    row[6511 + t] = s_so[10 + t * 2];                          // s_mu
    row[6521 + t] = 0.5f * sigmoidf_(fmaxf(s_so[10 + t * 2 + 1], -5.f)); // s_sig
    row[6531 + t] = s_om[t];                                   // om_logits
    float c0 = s_om[10 + t * 3], s0 = s_om[10 + t * 3 + 1], r2 = s_om[10 + t * 3 + 2];
    float kap = 7.f + expf(fminf(r2, 5.f));
    float nrm = sqrtf(c0 * c0 + s0 * s0) + 1e-3f;
    float cn = c0 / nrm, sn = s0 / nrm;
    float mu = asinf(fminf(fmaxf(sn, -1.f), 1.f));
    if (cn < 0.f) {
      float sg = (sn > 0.f) ? 1.f : ((sn < 0.f) ? -1.f : 0.f);
      mu = sg * PI_F - mu;
    }
    row[6541 + t] = cn; row[6551 + t] = sn;
    row[6561 + t] = kap; row[6571 + t] = mu;
  }
}

// ================================ launcher ===================================
extern "C" void kernel_launch(void* const* d_in, const int* in_sizes, int n_in,
                              void* d_out, int out_size, void* d_ws, size_t ws_size,
                              hipStream_t stream) {
  const float* output_f = (const float*)d_in[0];
  const float* map_f    = (const float*)d_in[1];
  const float* conv1_w  = (const float*)d_in[2];
  const float* conv1_b  = (const float*)d_in[3];
  const float* bn1_g    = (const float*)d_in[4];
  const float* bn1_b    = (const float*)d_in[5];
  const float* bn1_m    = (const float*)d_in[6];
  const float* bn1_v    = (const float*)d_in[7];
  const float* conv2_w  = (const float*)d_in[8];
  const float* conv2_b  = (const float*)d_in[9];
  const float* bn2_g    = (const float*)d_in[10];
  const float* bn2_b    = (const float*)d_in[11];
  const float* bn2_m    = (const float*)d_in[12];
  const float* bn2_v    = (const float*)d_in[13];
  const float* conv3_w  = (const float*)d_in[14];
  const float* conv3_b  = (const float*)d_in[15];
  const float* wl_w1 = (const float*)d_in[16]; const float* wl_b1 = (const float*)d_in[17];
  const float* wl_w2 = (const float*)d_in[18]; const float* wl_b2 = (const float*)d_in[19];
  const float* th_w1 = (const float*)d_in[20]; const float* th_b1 = (const float*)d_in[21];
  const float* th_w2 = (const float*)d_in[22]; const float* th_b2 = (const float*)d_in[23];
  const float* mv_w1 = (const float*)d_in[24]; const float* mv_b1 = (const float*)d_in[25];
  const float* mv_w2 = (const float*)d_in[26]; const float* mv_b2 = (const float*)d_in[27];
  const float* s_w1  = (const float*)d_in[28]; const float* s_b1  = (const float*)d_in[29];
  const float* s_w2  = (const float*)d_in[30]; const float* s_b2  = (const float*)d_in[31];
  const float* om_w1 = (const float*)d_in[32]; const float* om_b1 = (const float*)d_in[33];
  const float* om_w2 = (const float*)d_in[34]; const float* om_b2 = (const float*)d_in[35];

  char* ws = (char*)d_ws;
  _Float16* out_h = (_Float16*)(ws + WS_OUT_H);
  _Float16* wp1   = (_Float16*)(ws + WS_WP1);
  _Float16* wp2   = (_Float16*)(ws + WS_WP2);
  _Float16* x1h   = (_Float16*)(ws + WS_X1H);
  _Float16* x1p   = (_Float16*)(ws + WS_X1P);
  float*    lf    = (float*)(ws + WS_SMALL);
  float*    f2w   = lf + NB * 128;
  float*    out   = (float*)d_out;

  const int npack = NB * DM + 64 * K1 + 64 * K2;
  k_pack<<<(npack + 255) / 256, 256, 0, stream>>>(output_f, conv1_w, conv2_w,
                                                  out_h, wp1, wp2);

  dim3 cgrid(LPOS / 16, NB);
  k_conv1<<<cgrid, 64, 0, stream>>>(map_f, conv1_b, bn1_g, bn1_b, bn1_m, bn1_v,
                                    out_h, wp1, x1h);

  k_maxpool<<<(NB * 64 * LPOS + 255) / 256, 256, 0, stream>>>(x1h, x1p);

  k_conv2<<<cgrid, 64, 0, stream>>>(x1p, wp2, conv2_b, bn2_g, bn2_b, bn2_m, bn2_v,
                                    conv3_w, conv3_b, out);

  k_argmax<<<NB, 256, 0, stream>>>(out, map_f, lf);

  k_heads1<<<NB, 128, 0, stream>>>(output_f, lf, wl_w1, wl_b1, wl_w2, wl_b2,
                                   th_w1, th_b1, th_w2, th_b2, out, f2w);

  k_heads2<<<NB, 128, 0, stream>>>(f2w, mv_w1, mv_b1, mv_w2, mv_b2,
                                   s_w1, s_b1, s_w2, s_b2,
                                   om_w1, om_b1, om_w2, om_b2, out);
}
